// LocalFeatureAggregation_68856915689536
// MI455X (gfx1250) — matrix operand order repaired and checked by reading, not compiled, not run on hardware
//
#include <hip/hip_runtime.h>

#define NPOINTS 8192
#define NBATCH  2
#define KNN     16
#define TOTQ    (NBATCH * NPOINTS)   // 16384
#define BN_EPS  1e-5f

typedef __attribute__((ext_vector_type(16))) _Float16 v16h;
typedef __attribute__((ext_vector_type(8)))  float    v8f;

// ds_swizzle_b32 lane-xor exchange (group-of-32 mode):
// offset = (xor_mask << 10) | (or_mask << 5) | and_mask, and_mask=0x1F keeps
// all 5 lane bits, xor_mask flips lanes. Masks 1/2/4/8 stay within 16-lane
// halves (the WMMA C/D N-dimension). EXEC is all-ones where these run.
#define SWZ_XOR(v, m) \
    __int_as_float(__builtin_amdgcn_ds_swizzle(__float_as_int(v), (((m) << 10) | 0x1F)))

__device__ __forceinline__ float red_max16(float v)
{
    v = fmaxf(v, SWZ_XOR(v, 1));
    v = fmaxf(v, SWZ_XOR(v, 2));
    v = fmaxf(v, SWZ_XOR(v, 4));
    v = fmaxf(v, SWZ_XOR(v, 8));
    return v;
}
__device__ __forceinline__ float red_sum16(float v)
{
    v += SWZ_XOR(v, 1);
    v += SWZ_XOR(v, 2);
    v += SWZ_XOR(v, 4);
    v += SWZ_XOR(v, 8);
    return v;
}

// ---------------------------------------------------------------------------
// 1) Brute-force KNN (squared L2, excluding self), ascending order (k=16)
// ---------------------------------------------------------------------------
__global__ void __launch_bounds__(128)
k_knn(const float* __restrict__ coords, int* __restrict__ knn_idx,
      float* __restrict__ knn_dist)
{
    const int tid = threadIdx.x;
    const int q   = blockIdx.x * 128 + tid;       // global query (b*N+n)
    const int n   = q & (NPOINTS - 1);
    const int b   = q >> 13;
    const float cx = coords[q * 3 + 0];
    const float cy = coords[q * 3 + 1];
    const float cz = coords[q * 3 + 2];

    float best[KNN];
    int   bidx[KNN];
#pragma unroll
    for (int i = 0; i < KNN; ++i) { best[i] = 3.0e38f; bidx[i] = 0; }

    __shared__ float tile[128 * 3];
    const float* cb = coords + (size_t)b * NPOINTS * 3;

    for (int t = 0; t < NPOINTS / 128; ++t) {
        const int cand0 = t * 128;
        __syncthreads();
        tile[tid * 3 + 0] = cb[(cand0 + tid) * 3 + 0];
        tile[tid * 3 + 1] = cb[(cand0 + tid) * 3 + 1];
        tile[tid * 3 + 2] = cb[(cand0 + tid) * 3 + 2];
        if (t + 1 < NPOINTS / 128)
            __builtin_prefetch(&cb[(cand0 + 128 + tid) * 3], 0, 3); // global_prefetch_b8
        __syncthreads();
        for (int j = 0; j < 128; ++j) {
            const float dx = cx - tile[j * 3 + 0];
            const float dy = cy - tile[j * 3 + 1];
            const float dz = cz - tile[j * 3 + 2];
            const float d2 = dx * dx + dy * dy + dz * dz;
            const int cand = cand0 + j;
            if (cand != n && d2 < best[KNN - 1]) {
                // branch-free sorted insertion: compile-time indices only
#pragma unroll
                for (int s = KNN - 1; s >= 1; --s) {
                    const bool here = d2 < best[s];
                    const bool up   = d2 < best[s - 1];
                    if (here) {
                        best[s] = up ? best[s - 1] : d2;
                        bidx[s] = up ? bidx[s - 1] : cand;
                    }
                }
                if (d2 < best[0]) { best[0] = d2; bidx[0] = cand; }
            }
        }
    }
#pragma unroll
    for (int i = 0; i < KNN; ++i) {
        knn_idx [q * KNN + i] = bidx[i];
        knn_dist[q * KNN + i] = best[i];       // already >= 0 (direct form)
    }
}

// ---------------------------------------------------------------------------
// 2) Spatial features: [center(3), neighbor(3), rel(3), d2(1)] per (b,n,k)
// ---------------------------------------------------------------------------
__global__ void k_spatial(const float* __restrict__ coords,
                          const int* __restrict__ knn_idx,
                          const float* __restrict__ knn_dist,
                          float* __restrict__ spat)
{
    const int e = blockIdx.x * blockDim.x + threadIdx.x;   // (q,k) flat
    if (e >= TOTQ * KNN) return;
    const int q = e >> 4;
    const int b = q >> 13;
    const float cx = coords[q * 3 + 0], cy = coords[q * 3 + 1], cz = coords[q * 3 + 2];
    const int ni = knn_idx[e];
    const float* np = coords + ((size_t)(b * NPOINTS + ni)) * 3;
    const float nx = np[0], ny = np[1], nz = np[2];
    float* o = spat + (size_t)e * 10;
    o[0] = cx; o[1] = cy; o[2] = cz;
    o[3] = nx; o[4] = ny; o[5] = nz;
    o[6] = cx - nx; o[7] = cy - ny; o[8] = cz - nz;
    o[9] = knn_dist[e];
}

// ---------------------------------------------------------------------------
// 3) Pack pool weights (Ctot x Ctot f32) into f16 CDNA5 WMMA A-matrix layout.
//    A is 16x32 (MxK) per wmma: lane l -> row M=l%16, half h -> K within chunk
//    K(l,h) = 8*(l/16) + (h%8) + 16*(h/8).
// ---------------------------------------------------------------------------
__global__ void k_pack(const float* __restrict__ w, _Float16* __restrict__ packed,
                       int ctot)
{
    const int i = blockIdx.x * blockDim.x + threadIdx.x;
    const int ksn = ctot >> 5;
    const int total = (ctot / 16) * ksn * 512;
    if (i >= total) return;
    const int h    = i & 15;
    const int lane = (i >> 4) & 31;
    const int ks   = (i >> 9) % ksn;
    const int mt   = i / (512 * ksn);
    const int m    = mt * 16 + (lane & 15);
    const int kc   = ks * 32 + 8 * (lane >> 4) + (h & 7) + 16 * (h >> 3);
    packed[i] = (_Float16)w[m * ctot + kc];
}

// ---------------------------------------------------------------------------
// 4) Generic channel GEMM (1x1 conv): y[b,o,n] = sum_i W[o,i] x[b,i,n] + bias
// ---------------------------------------------------------------------------
__global__ void __launch_bounds__(256)
k_gemm(const float* __restrict__ x, const float* __restrict__ w,
       const float* __restrict__ bias, float* __restrict__ y,
       int cin, int cout)
{
    __shared__ float sw[8192];        // up to 128x64
    __shared__ float xt[128 * 32];    // cin x 32 columns
    __shared__ float sb[128];
    const int tid = threadIdx.x;
    for (int i = tid; i < cout * cin; i += 256) sw[i] = w[i];
    if (tid < cout) sb[tid] = bias[tid];
    const int col0 = blockIdx.x * 32;
    const int b  = col0 >> 13;
    const int n0 = col0 & (NPOINTS - 1);
    for (int i = tid; i < cin * 32; i += 256)
        xt[i] = x[((size_t)b * cin + (i >> 5)) * NPOINTS + n0 + (i & 31)];
    __syncthreads();
    for (int i = tid; i < cout * 32; i += 256) {
        const int o = i >> 5, c = i & 31;
        float acc = sb[o];
        for (int ci = 0; ci < cin; ++ci) acc += sw[o * cin + ci] * xt[(ci << 5) + c];
        y[((size_t)b * cout + o) * NPOINTS + n0 + c] = acc;
    }
}

// ---------------------------------------------------------------------------
// 5) Per-channel sum / sumsq over (B,N)  (one block per channel, no atomics)
// ---------------------------------------------------------------------------
__global__ void __launch_bounds__(256)
k_stats(const float* __restrict__ y, int cout,
        float* __restrict__ sums, float* __restrict__ sumsq)
{
    const int c = blockIdx.x, tid = threadIdx.x;
    float s = 0.f, s2 = 0.f;
    for (int i = tid; i < TOTQ; i += 256) {
        const int b = i >> 13, n = i & (NPOINTS - 1);
        const float v = y[((size_t)b * cout + c) * NPOINTS + n];
        s += v; s2 += v * v;
    }
    __shared__ float rs[256], rs2[256];
    rs[tid] = s; rs2[tid] = s2; __syncthreads();
    for (int w = 128; w > 0; w >>= 1) {
        if (tid < w) { rs[tid] += rs[tid + w]; rs2[tid] += rs2[tid + w]; }
        __syncthreads();
    }
    if (tid == 0) { sums[c] = rs[0]; sumsq[c] = rs2[0]; }
}

// Same, but for LSE pre-activation computed on the fly: y = W[c,:]·s + b[c]
__global__ void __launch_bounds__(256)
k_stats_lse(const float* __restrict__ spat, const float* __restrict__ w,
            const float* __restrict__ bias,
            float* __restrict__ sums, float* __restrict__ sumsq)
{
    const int c = blockIdx.x, tid = threadIdx.x;
    float wr[10];
#pragma unroll
    for (int j = 0; j < 10; ++j) wr[j] = w[c * 10 + j];
    const float bb = bias[c];
    float s = 0.f, s2 = 0.f;
    for (int e = tid; e < TOTQ * KNN; e += 256) {
        const float* sp = spat + (size_t)e * 10;
        float acc = bb;
#pragma unroll
        for (int j = 0; j < 10; ++j) acc += wr[j] * sp[j];
        s += acc; s2 += acc * acc;
    }
    __shared__ float rs[256], rs2[256];
    rs[tid] = s; rs2[tid] = s2; __syncthreads();
    for (int ww = 128; ww > 0; ww >>= 1) {
        if (tid < ww) { rs[tid] += rs[tid + ww]; rs2[tid] += rs2[tid + ww]; }
        __syncthreads();
    }
    if (tid == 0) { sums[c] = rs[0]; sumsq[c] = rs2[0]; }
}

// ---------------------------------------------------------------------------
// 6) BN finalize: scale = g*rsqrt(var+eps), shift = beta - mean*scale
// ---------------------------------------------------------------------------
__global__ void k_finalize(const float* __restrict__ sums, const float* __restrict__ sumsq,
                           const float* __restrict__ g, const float* __restrict__ bt,
                           int c, float invcnt,
                           float* __restrict__ scale, float* __restrict__ shift)
{
    const int i = threadIdx.x;
    if (i >= c) return;
    const float mean = sums[i] * invcnt;
    const float var  = sumsq[i] * invcnt - mean * mean;
    const float inv  = rsqrtf(var + BN_EPS);
    const float sc   = g[i] * inv;
    scale[i] = sc;
    shift[i] = bt[i] - mean * sc;
}

// ---------------------------------------------------------------------------
// 7) BN apply + activation (0=relu, 1=lrelu0.2, 2=lrelu0.01)
// ---------------------------------------------------------------------------
__global__ void k_apply(const float* __restrict__ y, const float* __restrict__ scale,
                        const float* __restrict__ shift, float* __restrict__ dst,
                        int cout, int dstC, int dstOff, int act)
{
    const int i = blockIdx.x * 256 + threadIdx.x;   // flat over (B,cout,N)
    const int n = i & (NPOINTS - 1);
    const int o = (i >> 13) % cout;
    const int b = i / (cout * NPOINTS);
    float v = y[i] * scale[o] + shift[o];
    if (act == 0)      v = fmaxf(v, 0.f);
    else if (act == 1) v = (v >= 0.f) ? v : 0.2f * v;
    else               v = (v >= 0.f) ? v : 0.01f * v;
    dst[((size_t)b * dstC + dstOff + o) * NPOINTS + n] = v;
}

// ---------------------------------------------------------------------------
// 8) Fused LSE + attentive pool (WMMA).  One wave per point.
//    F(Ctot x 16) = [ relu(bn(W_lse·s)), x broadcast ] in LDS (f16)
//    S(Ctot x 16) = W_pool · F + b   via v_wmma_f32_16x16x32_f16
//    softmax over K (16 lanes) via ds_swizzle xor butterflies, weighted sum
// ---------------------------------------------------------------------------
template <int CSF, int CFEAT>
__global__ void __launch_bounds__(128)
k_pool(const float* __restrict__ spat,
       const float* __restrict__ lse_w, const float* __restrict__ lse_b,
       const float* __restrict__ lse_scale, const float* __restrict__ lse_shift,
       const float* __restrict__ xfeat,          // (B, CFEAT, N)
       const _Float16* __restrict__ packedA,     // WMMA-A packed pool weights
       const float* __restrict__ pool_b,
       float* __restrict__ outp)                 // (B, CTOT, N)
{
    constexpr int CTOT = CSF + CFEAT;
    constexpr int MT   = CTOT / 16;   // M tiles of scores
    constexpr int KSN  = CTOT / 32;   // K steps of 32 channels

    __shared__ float sw[CSF * 10];
    __shared__ float sb[CSF], sscale[CSF], sshift[CSF];
    __shared__ float pbias[CTOT];
    __shared__ float sS[4][KNN * 10];
    __shared__ alignas(32) _Float16 Fh[4][CTOT * 16];

    const int tid  = threadIdx.x;
    const int wave = tid >> 5;
    const int lane = tid & 31;
    const int point = blockIdx.x * 4 + wave;     // exact cover of 16384 points
    const int b = point >> 13;
    const int n = point & (NPOINTS - 1);

    for (int i = tid; i < CSF * 10; i += 128) sw[i] = lse_w[i];
    for (int i = tid; i < CSF; i += 128) {
        sb[i] = lse_b[i]; sscale[i] = lse_scale[i]; sshift[i] = lse_shift[i];
    }
    for (int i = tid; i < CTOT; i += 128) pbias[i] = pool_b[i];

    // stage the 16 spatial rows (10 floats each) for this point
    if (lane < KNN) {
        const float* sp = spat + ((size_t)point * KNN + lane) * 10;
#pragma unroll
        for (int j = 0; j < 10; ++j) sS[wave][lane * 10 + j] = sp[j];
    }
    // broadcast feature rows of F (same value for all k)
    for (int i = lane; i < CFEAT * KNN; i += 32) {
        const int c = i >> 4, k = i & 15;
        Fh[wave][(CSF + c) * 16 + k] =
            (_Float16)xfeat[((size_t)b * CFEAT + c) * NPOINTS + n];
    }
    __syncthreads();
    // sf rows: relu(bn(W_lse·s + b))
    for (int i = lane; i < CSF * KNN; i += 32) {
        const int c = i >> 4, k = i & 15;
        float acc = sb[c];
        const float* srow = &sS[wave][k * 10];
#pragma unroll
        for (int j = 0; j < 10; ++j) acc += sw[c * 10 + j] * srow[j];
        acc = acc * sscale[c] + sshift[c];
        acc = fmaxf(acc, 0.f);
        Fh[wave][c * 16 + k] = (_Float16)acc;
    }
    __syncthreads();

    const int halfsel = lane >> 4;     // C/D layout: M = r + 8*(lane/16)
    const int ncol    = lane & 15;     // C/D layout: N = lane % 16
#pragma unroll
    for (int mt = 0; mt < MT; ++mt) {
        v8f acc = {};
#pragma unroll
        for (int ks = 0; ks < KSN; ++ks) {
            // A: packed W_pool tile; B: 32 channel-rows of F, 16 neighbor cols
            v16h a  = *(const v16h*)(packedA + ((size_t)(mt * KSN + ks) * 32 + lane) * 16);
            v16h bm;
#pragma unroll
            for (int s = 0; s < 16; ++s)
                bm[s] = Fh[wave][(ks * 32 + 8 * halfsel + (s & 7) + 16 * (s >> 3)) * 16 + ncol];
            acc = __builtin_amdgcn_wmma_f32_16x16x32_f16(
                false, a, false, bm, (short)0, acc, false, false);
        }
#pragma unroll
        for (int r = 0; r < 8; ++r) {
            const int m = mt * 16 + r + 8 * halfsel;
            const float sv = acc[r] + pbias[m];
            const float mx = red_max16(sv);        // softmax over the 16 K-cols
            const float e  = __expf(sv - mx);
            const float ssum = red_sum16(e);
            const float aw = e / ssum;
            const float wv = red_sum16(aw * (float)Fh[wave][m * 16 + ncol]);
            if (ncol == 0)
                outp[((size_t)b * CTOT + m) * NPOINTS + n] = wv;
        }
    }
}

// ---------------------------------------------------------------------------
// Host orchestration
// ---------------------------------------------------------------------------
extern "C" void kernel_launch(void* const* d_in, const int* in_sizes, int n_in,
                              void* d_out, int out_size, void* d_ws, size_t ws_size,
                              hipStream_t stream)
{
    (void)in_sizes; (void)n_in; (void)out_size; (void)ws_size;
    const float* coords   = (const float*)d_in[0];
    const float* features = (const float*)d_in[1];
    const float *mlp1_w = (const float*)d_in[2],  *mlp1_b = (const float*)d_in[3],
                *mlp1_g = (const float*)d_in[4],  *mlp1_bt = (const float*)d_in[5];
    const float *lse1_w = (const float*)d_in[6],  *lse1_b = (const float*)d_in[7],
                *lse1_g = (const float*)d_in[8],  *lse1_bt = (const float*)d_in[9];
    const float *mlpp1_w = (const float*)d_in[10], *mlpp1_b = (const float*)d_in[11],
                *mlpp1_g = (const float*)d_in[12], *mlpp1_bt = (const float*)d_in[13];
    const float *lse2_w = (const float*)d_in[14], *lse2_b = (const float*)d_in[15],
                *lse2_g = (const float*)d_in[16], *lse2_bt = (const float*)d_in[17];
    const float *mlp2_w = (const float*)d_in[18], *mlp2_b = (const float*)d_in[19],
                *mlp2_g = (const float*)d_in[20], *mlp2_bt = (const float*)d_in[21];
    const float *res_w  = (const float*)d_in[22], *res_b  = (const float*)d_in[23],
                *res_g  = (const float*)d_in[24], *res_bt = (const float*)d_in[25];
    const float *pool1_w = (const float*)d_in[26], *pool1_b = (const float*)d_in[27];
    const float *pool2_w = (const float*)d_in[28], *pool2_b = (const float*)d_in[29];
    float* out = (float*)d_out;

    char* base = (char*)d_ws;
    size_t off = 0;
    auto alloc = [&](size_t bytes) -> char* {
        char* p = base + off;
        off += (bytes + 255) & ~(size_t)255;
        return p;
    };
    int*      knn_idx  = (int*)     alloc((size_t)TOTQ * KNN * 4);
    float*    knn_dist = (float*)   alloc((size_t)TOTQ * KNN * 4);
    float*    spat     = (float*)   alloc((size_t)TOTQ * KNN * 10 * 4);
    float*    ya       = (float*)   alloc((size_t)NBATCH * 64  * NPOINTS * 4); // mlp1 y / x1
    float*    p1       = (float*)   alloc((size_t)NBATCH * 128 * NPOINTS * 4);
    float*    yb       = (float*)   alloc((size_t)NBATCH * 32  * NPOINTS * 4); // mlpp1 y / x2
    float*    p2       = (float*)   alloc((size_t)NBATCH * 64  * NPOINTS * 4);
    float*    yc       = (float*)   alloc((size_t)NBATCH * 128 * NPOINTS * 4); // mlp2 y
    float*    yd       = (float*)   alloc((size_t)NBATCH * 128 * NPOINTS * 4); // res y
    _Float16* packA1   = (_Float16*)alloc((size_t)8 * 4 * 512 * 2);
    _Float16* packA2   = (_Float16*)alloc((size_t)4 * 2 * 512 * 2);
    float*    st       = (float*)   alloc((size_t)6 * 512 * 4);
    auto SUM = [&](int l) { return st + l * 512; };
    auto SQ  = [&](int l) { return st + l * 512 + 128; };
    auto SC  = [&](int l) { return st + l * 512 + 256; };
    auto SH  = [&](int l) { return st + l * 512 + 384; };

    const float invQ = 1.f / (float)TOTQ;
    const float invE = 1.f / (float)(TOTQ * KNN);

    // KNN + spatial features (shared by lse1 & lse2)
    k_knn<<<TOTQ / 128, 128, 0, stream>>>(coords, knn_idx, knn_dist);
    k_spatial<<<(TOTQ * KNN) / 256, 256, 0, stream>>>(coords, knn_idx, knn_dist, spat);

    // pack pool weights to f16 WMMA-A layout
    k_pack<<<(8 * 4 * 512) / 256, 256, 0, stream>>>(pool1_w, packA1, 128);
    k_pack<<<(4 * 2 * 512) / 256, 256, 0, stream>>>(pool2_w, packA2, 64);

    // mlp1: 32 -> 64, BN, LeakyReLU(0.2)
    k_gemm<<<TOTQ / 32, 256, 0, stream>>>(features, mlp1_w, mlp1_b, ya, 32, 64);
    k_stats<<<64, 256, 0, stream>>>(ya, 64, SUM(0), SQ(0));
    k_finalize<<<1, 64, 0, stream>>>(SUM(0), SQ(0), mlp1_g, mlp1_bt, 64, invQ, SC(0), SH(0));
    k_apply<<<(NBATCH * 64 * NPOINTS) / 256, 256, 0, stream>>>(ya, SC(0), SH(0), ya, 64, 64, 0, 1);

    // lse1 BN stats (on the fly), then fused LSE1 + attentive pool 1 (WMMA)
    k_stats_lse<<<64, 256, 0, stream>>>(spat, lse1_w, lse1_b, SUM(1), SQ(1));
    k_finalize<<<1, 64, 0, stream>>>(SUM(1), SQ(1), lse1_g, lse1_bt, 64, invE, SC(1), SH(1));
    k_pool<64, 64><<<TOTQ / 4, 128, 0, stream>>>(spat, lse1_w, lse1_b, SC(1), SH(1),
                                                 ya, packA1, pool1_b, p1);

    // mlpp1: 128 -> 32, BN, ReLU
    k_gemm<<<TOTQ / 32, 256, 0, stream>>>(p1, mlpp1_w, mlpp1_b, yb, 128, 32);
    k_stats<<<32, 256, 0, stream>>>(yb, 32, SUM(2), SQ(2));
    k_finalize<<<1, 32, 0, stream>>>(SUM(2), SQ(2), mlpp1_g, mlpp1_bt, 32, invQ, SC(2), SH(2));
    k_apply<<<(NBATCH * 32 * NPOINTS) / 256, 256, 0, stream>>>(yb, SC(2), SH(2), yb, 32, 32, 0, 0);

    // lse2 BN stats, then fused LSE2 + attentive pool 2 (WMMA)
    k_stats_lse<<<32, 256, 0, stream>>>(spat, lse2_w, lse2_b, SUM(3), SQ(3));
    k_finalize<<<1, 32, 0, stream>>>(SUM(3), SQ(3), lse2_g, lse2_bt, 32, invE, SC(3), SH(3));
    k_pool<32, 32><<<TOTQ / 4, 128, 0, stream>>>(spat, lse2_w, lse2_b, SC(3), SH(3),
                                                 yb, packA2, pool2_b, p2);

    // mlp2: 64 -> 128, BN, ReLU -> out channels [0,128). Final LeakyReLU(0.01)
    // after ReLU is the identity, so plain ReLU is exact.
    k_gemm<<<TOTQ / 32, 256, 0, stream>>>(p2, mlp2_w, mlp2_b, yc, 64, 128);
    k_stats<<<128, 256, 0, stream>>>(yc, 128, SUM(4), SQ(4));
    k_finalize<<<1, 128, 0, stream>>>(SUM(4), SQ(4), mlp2_g, mlp2_bt, 128, invQ, SC(4), SH(4));
    k_apply<<<(NBATCH * 128 * NPOINTS) / 256, 256, 0, stream>>>(yc, SC(4), SH(4), out, 128, 256, 0, 0);

    // residual: 32 -> 128, BN, ReLU -> out channels [128,256)
    k_gemm<<<TOTQ / 32, 256, 0, stream>>>(features, res_w, res_b, yd, 32, 128);
    k_stats<<<128, 256, 0, stream>>>(yd, 128, SUM(5), SQ(5));
    k_finalize<<<1, 128, 0, stream>>>(SUM(5), SQ(5), res_g, res_bt, 128, invQ, SC(5), SH(5));
    k_apply<<<(NBATCH * 128 * NPOINTS) / 256, 256, 0, stream>>>(yd, SC(5), SH(5), out, 128, 256, 128, 0);
}
